// MultiScaleAttention_80676665688258
// MI455X (gfx1250) — matrix operand order repaired and checked by reading, not compiled, not run on hardware
//
#include <hip/hip_runtime.h>
#include <hip/hip_bf16.h>
#include <stdint.h>

typedef __bf16 bf16;
typedef __attribute__((ext_vector_type(16))) __bf16 v16bf;
typedef __attribute__((ext_vector_type(8)))  __bf16 v8bf;
typedef __attribute__((ext_vector_type(8)))  float  v8f;

#define B_     2
#define NQ_    4096        // N*S per batch
#define M_     2048
#define H_     8
#define D_     64
#define QD_    512
#define INNER_ 512
#define ROWS_  (B_ * NQ_)  // 8192
#define CROWS_ (B_ * M_)   // 4096

#define WMMA_BF16(av, bv, cv) \
    __builtin_amdgcn_wmma_f32_16x16x32_bf16(false, (av), false, (bv), (short)0, (cv), false, false)

#define SHUF16(a, b) __builtin_shufflevector((a), (b), 0,1,2,3,4,5,6,7,8,9,10,11,12,13,14,15)

// ---------------- async global -> LDS copy (16 bytes), gfx1250 path ----------------
#if __has_builtin(__builtin_amdgcn_global_load_async_to_lds_b128)
#define ASYNC_OK 1
typedef int v4i_vs __attribute__((vector_size(16)));
typedef __attribute__((address_space(1))) v4i_vs as1_v4i;
typedef __attribute__((address_space(3))) v4i_vs as3_v4i;
__device__ __forceinline__ void async_cp16(const bf16* g, bf16* l) {
    __builtin_amdgcn_global_load_async_to_lds_b128(
        (as1_v4i*)(unsigned long long)(uintptr_t)g,
        (as3_v4i*)(unsigned)(uintptr_t)l, 0, 0);
}
__device__ __forceinline__ void wait_async() {
    asm volatile("s_wait_asynccnt 0" ::: "memory");
}
#else
#define ASYNC_OK 0
__device__ __forceinline__ void async_cp16(const bf16* g, bf16* l) {
    *(v8bf*)l = *(const v8bf*)g;   // sync fallback: load + ds_store
}
__device__ __forceinline__ void wait_async() {}
#endif

// ---------------- DPP ROW_XMASK butterfly reductions within 16-lane rows ----------------
template <int XM>
__device__ __forceinline__ float dpp_xor_max(float x) {
    int y = __builtin_amdgcn_update_dpp(0, __builtin_bit_cast(int, x),
                                        0x160 + XM, 0xf, 0xf, true);
    return fmaxf(x, __builtin_bit_cast(float, y));
}
template <int XM>
__device__ __forceinline__ float dpp_xor_add(float x) {
    int y = __builtin_amdgcn_update_dpp(0, __builtin_bit_cast(int, x),
                                        0x160 + XM, 0xf, 0xf, true);
    return x + __builtin_bit_cast(float, y);
}
__device__ __forceinline__ float row16_max(float x) {
    return dpp_xor_max<8>(dpp_xor_max<4>(dpp_xor_max<2>(dpp_xor_max<1>(x))));
}
__device__ __forceinline__ float row16_sum(float x) {
    return dpp_xor_add<8>(dpp_xor_add<4>(dpp_xor_add<2>(dpp_xor_add<1>(x))));
}

// ---------------- convert f32 -> bf16 (grid-stride) ----------------
__global__ void k_cvt(const float* __restrict__ src, bf16* __restrict__ dst, int n) {
    int i = blockIdx.x * blockDim.x + threadIdx.x;
    int stride = gridDim.x * blockDim.x;
    for (; i < n; i += stride) dst[i] = (bf16)src[i];
}

// -------- transpose + convert: dst[o][i] = src[i][col_off + o] --------
__global__ void k_transpose_cvt(const float* __restrict__ src, bf16* __restrict__ dst,
                                int rows, int cols, int src_ld, int col_off) {
    int o = blockIdx.x * 16 + threadIdx.x;
    int i = blockIdx.y * 16 + threadIdx.y;
    if (o < cols && i < rows)
        dst[(size_t)o * rows + i] = (bf16)src[(size_t)i * src_ld + col_off + o];
}

// -------- V re-layout: vt[(b,h)][d][m] = vpart[(b*M+m)][h*64+d] --------
__global__ void k_vtrans(const bf16* __restrict__ vpart, bf16* __restrict__ vt, int n) {
    int t = blockIdx.x * blockDim.x + threadIdx.x;
    int stride = gridDim.x * blockDim.x;
    for (; t < n; t += stride) {
        int m = t & (M_ - 1);
        int rest = t >> 11;
        int d = rest & (D_ - 1);
        int bh = rest >> 6;
        int b = bh >> 3, h = bh & 7;
        vt[t] = vpart[(size_t)(b * M_ + m) * INNER_ + h * D_ + d];
    }
}

// ---------------- WMMA GEMM: C = A @ B, BT[N][K] given transposed ----------------
// block 128 = 4 waves; wave computes 16 rows x 64 cols (4 WMMAs / k-step, A reused x4).
// B k-slab (64 cols x 32 k = 4KB) staged via async global->LDS, double buffered,
// shared by all 4 waves.
template <bool FP32OUT>
__global__ void __launch_bounds__(128, 1)
k_gemm(const bf16* __restrict__ A, const bf16* __restrict__ BT,
       void* __restrict__ Cout, const float* __restrict__ bias,
       int Nc, int Kd) {
    __shared__ __align__(32) bf16 Bs[2][64 * 32];

    const int tid = threadIdx.x;
    const int lane = tid & 31, wave = tid >> 5;
    const int hl = lane >> 4, l15 = lane & 15;
    const int row0 = blockIdx.y * 64 + wave * 16;
    const int col0 = blockIdx.x * 64;

    const bf16* arow = A + (size_t)(row0 + l15) * Kd + hl * 16;

    // staging: 128 threads x 32B = 4KB slab
    const int ci = tid >> 1;            // B-row 0..63
    const int ch = (tid & 1) * 16;      // element offset within 32-wide k slab
    const bf16* bsrc = BT + (size_t)(col0 + ci) * Kd + ch;
    bf16* bd0 = &Bs[0][ci * 32 + ch];
    bf16* bd1 = &Bs[1][ci * 32 + ch];

    async_cp16(bsrc, bd0);
    async_cp16(bsrc + 8, bd0 + 8);

    v8f acc[4] = {v8f{}, v8f{}, v8f{}, v8f{}};
    int buf = 0;
    for (int k = 0; k < Kd; k += 32) {
        wait_async();
        asm volatile("s_wait_dscnt 0" ::: "memory");
        __syncthreads();
        if (k + 32 < Kd) {
            bf16* nd = buf ? bd0 : bd1;
            async_cp16(bsrc + k + 32, nd);
            async_cp16(bsrc + k + 40, nd + 8);
        }
        v8bf a0 = *(const v8bf*)(arow + k);
        v8bf a1 = *(const v8bf*)(arow + k + 8);
        v16bf av = SHUF16(a0, a1);
        const bf16* bb = &Bs[buf][l15 * 32 + hl * 16];
        for (int t = 0; t < 4; ++t) {
            v16bf bv = *(const v16bf*)(bb + t * 16 * 32);
            acc[t] = WMMA_BF16(av, bv, acc[t]);
        }
        buf ^= 1;
    }

    for (int t = 0; t < 4; ++t) {
        int c = col0 + t * 16 + l15;
        float badd = FP32OUT ? bias[c] : 0.0f;
        for (int j = 0; j < 8; ++j) {
            int r = row0 + hl * 8 + j;
            if (FP32OUT)
                ((float*)Cout)[(size_t)r * Nc + c] = acc[t][j] + badd;
            else
                ((bf16*)Cout)[(size_t)r * Nc + c] = (bf16)acc[t][j];
        }
    }
}

// ---------------- fused flash attention ----------------
// grid (NQ_/64, B*H), block 128 (4 waves). Wave = 16 queries x d=64.
// K/V 32-key slabs staged via async global->LDS (double buffered, shared by 4 waves).
__global__ void __launch_bounds__(128, 1)
k_attn(const bf16* __restrict__ Q,   // [ROWS_][512], head col h*64
       const bf16* __restrict__ Kp,  // [CROWS_][512], head col h*64
       const bf16* __restrict__ Vt,  // [(b,h)*64 + d][M_]
       bf16* __restrict__ O) {       // [ROWS_][512]
    __shared__ __align__(32) bf16 Ks[2][32 * 64];    // [m-local][d]
    __shared__ __align__(32) bf16 Vs[2][64 * 32];    // [d][m-local]
    __shared__ __align__(32) bf16 plds[4][16 * 32];  // per-wave P transpose buffer

    const int tid = threadIdx.x;
    const int bh = blockIdx.y, b = bh >> 3, h = bh & 7;
    const int lane = tid & 31, wave = tid >> 5;
    const int hl = lane >> 4, l15 = lane & 15;
    const int q0 = (blockIdx.x * 4 + wave) * 16;
    const float scale = 0.125f;  // 64^-0.5

    // persistent Q fragments
    const bf16* qrow = Q + (size_t)(b * NQ_ + q0 + l15) * INNER_ + h * D_ + hl * 16;
    v16bf aQ[2];
    for (int kk = 0; kk < 2; ++kk) {
        v8bf a0 = *(const v8bf*)(qrow + kk * 32);
        v8bf a1 = *(const v8bf*)(qrow + kk * 32 + 8);
        aQ[kk] = SHUF16(a0, a1);
    }

    const bf16* kbase = Kp + (size_t)(b * M_) * INNER_ + h * D_;
    const bf16* vbase = Vt + (size_t)bh * D_ * M_;

    // staging indices: K slab 32x64 (4KB), V slab 64x32 (4KB); 64B per thread
    const int ki = tid >> 2, kq = (tid & 3) * 16;
    const int vi = tid >> 1, vq = (tid & 1) * 16;
    const bf16* ksrc = kbase + (size_t)ki * INNER_ + kq;
    const bf16* vsrc = vbase + (size_t)vi * M_ + vq;

    auto stage = [&](int bufi, int mb) {
        const bf16* kp = ksrc + (size_t)mb * INNER_;
        bf16* kd = &Ks[bufi][ki * 64 + kq];
        async_cp16(kp, kd);
        async_cp16(kp + 8, kd + 8);
        const bf16* vp = vsrc + mb;
        bf16* vd = &Vs[bufi][vi * 32 + vq];
        async_cp16(vp, vd);
        async_cp16(vp + 8, vd + 8);
    };

    v8f acc0 = {}, acc1 = {}, acc2 = {}, acc3 = {};
    float mrow[8], lrow[8];
    for (int j = 0; j < 8; ++j) { mrow[j] = -1e30f; lrow[j] = 0.0f; }

    bf16* myl = &plds[wave][0];
    stage(0, 0);
    int buf = 0;

    for (int mb = 0; mb < M_; mb += 32) {
        wait_async();
        asm volatile("s_wait_dscnt 0" ::: "memory");
        __syncthreads();
        if (mb + 32 < M_) stage(buf ^ 1, mb + 32);

        // ---- scores: 16 q x 32 keys ----
        v8f s[2];
        for (int nt = 0; nt < 2; ++nt) {
            v8f sa = {};
            const bf16* kr = &Ks[buf][(nt * 16 + l15) * 64 + hl * 16];
            for (int kk = 0; kk < 2; ++kk) {
                v16bf bv = *(const v16bf*)(kr + kk * 32);
                sa = WMMA_BF16(aQ[kk], bv, sa);
            }
            for (int j = 0; j < 8; ++j) s[nt][j] = sa[j] * scale;
        }

        // ---- online softmax (DPP row reductions, no LDS) ----
        float tmax[8], rsum[8];
        for (int j = 0; j < 8; ++j)
            tmax[j] = row16_max(fmaxf(s[0][j], s[1][j]));
        for (int j = 0; j < 8; ++j) {
            float nm = fmaxf(mrow[j], tmax[j]);
            float corr = __expf(mrow[j] - nm);
            mrow[j] = nm;
            lrow[j] *= corr;
            acc0[j] *= corr; acc1[j] *= corr; acc2[j] *= corr; acc3[j] *= corr;
        }

        // ---- P tile -> per-wave LDS (C-layout -> memory row-major) ----
        asm volatile("" ::: "memory");
        for (int j = 0; j < 8; ++j) {
            float p0 = __expf(s[0][j] - mrow[j]);
            float p1 = __expf(s[1][j] - mrow[j]);
            rsum[j] = p0 + p1;
            int r = hl * 8 + j;
            myl[r * 32 + l15]      = (bf16)p0;
            myl[r * 32 + 16 + l15] = (bf16)p1;
        }
        for (int j = 0; j < 8; ++j) lrow[j] += row16_sum(rsum[j]);

        // wave-internal LDS RAW fence (DS in-order per wave)
        asm volatile("s_wait_dscnt 0" ::: "memory");

        // ---- reload P as A-fragment ----
        const bf16* pl = myl + l15 * 32 + hl * 16;
        v8bf p0v = *(const v8bf*)(pl);
        v8bf p1v = *(const v8bf*)(pl + 8);
        v16bf pa = SHUF16(p0v, p1v);

        // ---- out += P @ V (B-frags from LDS V slab) ----
        const bf16* vr = &Vs[buf][l15 * 32 + hl * 16];
        acc0 = WMMA_BF16(pa, *(const v16bf*)(vr + 0 * 16 * 32), acc0);
        acc1 = WMMA_BF16(pa, *(const v16bf*)(vr + 1 * 16 * 32), acc1);
        acc2 = WMMA_BF16(pa, *(const v16bf*)(vr + 2 * 16 * 32), acc2);
        acc3 = WMMA_BF16(pa, *(const v16bf*)(vr + 3 * 16 * 32), acc3);
        asm volatile("" ::: "memory");
        buf ^= 1;
    }

    // ---- epilogue ----
    for (int j = 0; j < 8; ++j) {
        float inv = 1.0f / lrow[j];
        size_t r = (size_t)(b * NQ_ + q0 + hl * 8 + j);
        bf16* op = O + r * INNER_ + h * D_ + l15;
        op[0]  = (bf16)(acc0[j] * inv);
        op[16] = (bf16)(acc1[j] * inv);
        op[32] = (bf16)(acc2[j] * inv);
        op[48] = (bf16)(acc3[j] * inv);
    }
}

// ---------------- host-side orchestration ----------------
extern "C" void kernel_launch(void* const* d_in, const int* in_sizes, int n_in,
                              void* d_out, int out_size, void* d_ws, size_t ws_size,
                              hipStream_t stream) {
    const float* x   = (const float*)d_in[0];
    const float* ctx = (const float*)d_in[1];
    const float* Wq  = (const float*)d_in[2];
    const float* Wkv = (const float*)d_in[3];
    const float* Wo  = (const float*)d_in[4];
    const float* bo  = (const float*)d_in[5];
    float* out = (float*)d_out;

    char* p = (char*)d_ws;
    auto alloc = [&](size_t nelem) {
        bf16* r = (bf16*)p;
        size_t bytes = (nelem * sizeof(bf16) + 255) & ~(size_t)255;
        p += bytes;
        return r;
    };
    bf16* xb    = alloc((size_t)ROWS_ * QD_);
    bf16* cb    = alloc((size_t)CROWS_ * QD_);
    bf16* WqT   = alloc((size_t)INNER_ * QD_);
    bf16* WkT   = alloc((size_t)INNER_ * QD_);
    bf16* WvT   = alloc((size_t)INNER_ * QD_);
    bf16* WoT   = alloc((size_t)QD_ * INNER_);
    bf16* qb    = alloc((size_t)ROWS_ * INNER_);
    bf16* kpart = alloc((size_t)CROWS_ * INNER_);
    bf16* vpart = alloc((size_t)CROWS_ * INNER_);
    bf16* vtb   = alloc((size_t)B_ * H_ * D_ * M_);
    bf16* aob   = alloc((size_t)ROWS_ * INNER_);

    // 1) conversions
    k_cvt<<<2048, 256, 0, stream>>>(x,   xb, ROWS_  * QD_);
    k_cvt<<<1024, 256, 0, stream>>>(ctx, cb, CROWS_ * QD_);
    dim3 tb(16, 16);
    k_transpose_cvt<<<dim3(32, 32), tb, 0, stream>>>(Wq,  WqT, QD_, INNER_, INNER_,     0);
    k_transpose_cvt<<<dim3(32, 32), tb, 0, stream>>>(Wkv, WkT, QD_, INNER_, 2 * INNER_, 0);
    k_transpose_cvt<<<dim3(32, 32), tb, 0, stream>>>(Wkv, WvT, QD_, INNER_, 2 * INNER_, INNER_);
    k_transpose_cvt<<<dim3(32, 32), tb, 0, stream>>>(Wo,  WoT, INNER_, QD_, QD_,        0);

    // 2) projections
    k_gemm<false><<<dim3(INNER_ / 64, ROWS_ / 64),  128, 0, stream>>>(xb, WqT, qb,    nullptr, INNER_, QD_);
    k_gemm<false><<<dim3(INNER_ / 64, CROWS_ / 64), 128, 0, stream>>>(cb, WkT, kpart, nullptr, INNER_, QD_);
    k_gemm<false><<<dim3(INNER_ / 64, CROWS_ / 64), 128, 0, stream>>>(cb, WvT, vpart, nullptr, INNER_, QD_);

    // 3) V -> [bh][d][m]
    k_vtrans<<<1024, 256, 0, stream>>>(vpart, vtb, B_ * H_ * D_ * M_);

    // 4) fused attention
    k_attn<<<dim3(NQ_ / 64, B_ * H_), 128, 0, stream>>>(qb, kpart, vtb, aob);

    // 5) output projection (f32 + bias)
    k_gemm<true><<<dim3(QD_ / 64, ROWS_ / 64), 128, 0, stream>>>(aob, WoT, out, bo, QD_, INNER_);
}
